// P2PNetPointnet2_23484881175034
// MI455X (gfx1250) — compile-verified, matrix operand order checked by reading, not run on hardware
//
#include <hip/hip_runtime.h>
#include <hip/hip_bf16.h>

// ---------------------------------------------------------------------------
// CDNA5 (gfx1250) implementation of P2PNet / PointNet++ forward.
// Matmuls -> V_WMMA_F32_16X16X32_F16 (f16 in, f32 accumulate).
// ---------------------------------------------------------------------------

typedef __attribute__((ext_vector_type(16))) _Float16 v16h;
typedef __attribute__((ext_vector_type(8)))  _Float16 v8h;
typedef __attribute__((ext_vector_type(8)))  float    v8f;

#define EPI_NONE   0
#define EPI_LNRELU 1
#define EPI_LN     2
#define EPI_SIG    3

#define BIGF 3.0e38f

// ---------------------------------------------------------------------------
// Fused GEMM: out[M,N] = A[M,Kp] @ Wt[N,Kp]^T + bias, epilogue in LDS tile.
// One workgroup = 16 output rows x full N (8 waves sweep column tiles), so the
// per-row LayerNorm can be fused right after the WMMA accumulation.
// A-fragment layout (16-bit A 16x32, wave32):
//   lanes 0-15  : row = lane,    halves 0-7 = K+0..7,  halves 8-15 = K+16..23
//   lanes 16-31 : row = lane-16, halves 0-7 = K+8..15, halves 8-15 = K+24..31
// B (Wt) is stored column-major-of-W (i.e. [n][k]) so it loads identically.
// C/D layout: VGPR r holds row r (lanes 0-15) / row r+8 (lanes 16-31), col=lane&15.
// ---------------------------------------------------------------------------
__global__ __launch_bounds__(256) void lin_kernel(
    const _Float16* __restrict__ A, int Kp,
    const _Float16* __restrict__ Wt,
    const float* __restrict__ bias,
    const float* __restrict__ gamma,
    const float* __restrict__ beta,
    _Float16* __restrict__ outH,
    float* __restrict__ outF,
    int N, int Nout, int epi)
{
  extern __shared__ float tile[];                 // 16 * N floats
  const int wave = threadIdx.x >> 5;
  const int lane = threadIdx.x & 31;
  const int hl   = lane & 15;
  const int hi   = lane >> 4;
  const long long rowBase = (long long)blockIdx.x * 16;
  const int nCT = N >> 4;

  for (int ct = wave; ct < nCT; ct += 8) {
    const _Float16* ar = A  + (rowBase + hl) * (long long)Kp;
    const _Float16* br = Wt + (long long)(ct * 16 + hl) * Kp;
    __builtin_prefetch(br + Kp, 0, 1);            // global_prefetch_b8
    v8f acc = {};
    for (int kk = 0; kk < Kp; kk += 32) {
      union { v16h v; v8h h[2]; } a, w;
      const int o0 = kk + hi * 8;
      const int o1 = kk + 16 + hi * 8;
      a.h[0] = *(const v8h*)(ar + o0);
      a.h[1] = *(const v8h*)(ar + o1);
      w.h[0] = *(const v8h*)(br + o0);
      w.h[1] = *(const v8h*)(br + o1);
      acc = __builtin_amdgcn_wmma_f32_16x16x32_f16(false, a.v, false, w.v,
                                                   (short)0, acc, false, false);
    }
    const int cc = ct * 16 + hl;
    const float bv = (cc < Nout) ? bias[cc] : 0.f;
#pragma unroll
    for (int r = 0; r < 8; r++)
      tile[(r + hi * 8) * N + cc] = acc[r] + bv;
  }
  __syncthreads();

  if (epi == EPI_LNRELU || epi == EPI_LN) {
    // 8 waves x 2 rows each = 16 rows; wave32 shfl reduction over N columns.
    for (int rr = 0; rr < 2; rr++) {
      const int row = wave * 2 + rr;
      float s = 0.f, s2 = 0.f;
      for (int c = lane; c < N; c += 32) {
        float v = tile[row * N + c];
        s += v; s2 += v * v;
      }
      for (int m = 16; m; m >>= 1) {
        s  += __shfl_xor(s,  m, 32);
        s2 += __shfl_xor(s2, m, 32);
      }
      const float mean = s / (float)N;
      const float var  = s2 / (float)N - mean * mean;
      const float rstd = rsqrtf(var + 1e-5f);
      for (int c = lane; c < N; c += 32) {
        float v = (tile[row * N + c] - mean) * rstd * gamma[c] + beta[c];
        if (epi == EPI_LNRELU) v = fmaxf(v, 0.f);
        outH[(rowBase + row) * N + c] = (_Float16)v;
      }
    }
  } else {
    for (int t = threadIdx.x; t < 16 * N; t += 256) {
      const int row = t / N, c = t - row * N;
      const float v = tile[t];
      if (epi == EPI_SIG) {
        if (c < Nout) outF[(rowBase + row) * Nout + c] = tanhf(0.5f * v); // sigmoid*2-1
      } else {
        if (outH) outH[(rowBase + row) * N + c] = (_Float16)v;
        if (outF) outF[(rowBase + row) * Nout + c] = v;   // Nout == N here
      }
    }
  }
}

// Convert fp32 W[Kin x Nreal] (row-major) -> f16 Wt[Npad][Kp], zero padded.
__global__ void convert_wt(const float* __restrict__ W, int Kin, int Nreal,
                           int Npad, int Kp, _Float16* __restrict__ Wt)
{
  int t = blockIdx.x * 256 + threadIdx.x;
  if (t >= Npad * Kp) return;
  int n = t / Kp, k = t - n * Kp;
  float v = (n < Nreal && k < Kin) ? W[(size_t)k * Nreal + n] : 0.f;
  Wt[t] = (_Float16)v;
}

// ---------------------------------------------------------------------------
// Farthest point sampling: one workgroup per batch, sequential by nature.
// idx[0]=0; step s: dmin=min(dmin, d(.,pos[last])); last=argmax(dmin) (ties->lowest).
// ---------------------------------------------------------------------------
__global__ __launch_bounds__(256) void fps_kernel(const float* __restrict__ pos,
                                                  int Ns, int nS, int* __restrict__ idx)
{
  __shared__ float dmin[6144];
  __shared__ float rv[256];
  __shared__ int   ri[256];
  __shared__ int   s_last;
  const int b = blockIdx.x;
  const int tid = threadIdx.x;
  const float* P = pos + (long long)b * Ns * 3;
  for (int i = tid; i < Ns; i += 256) dmin[i] = BIGF;
  if (tid == 0) { idx[b * nS] = 0; s_last = 0; }
  __syncthreads();
  for (int s = 1; s < nS; s++) {
    const int last = s_last;
    const float cx = P[last * 3 + 0], cy = P[last * 3 + 1], cz = P[last * 3 + 2];
    float mv = -1.f; int mi = 0;
    for (int i = tid; i < Ns; i += 256) {
      float dx = P[i * 3 + 0] - cx, dy = P[i * 3 + 1] - cy, dz = P[i * 3 + 2] - cz;
      float dm = fminf(dmin[i], dx * dx + dy * dy + dz * dz);
      dmin[i] = dm;
      if (dm > mv) { mv = dm; mi = i; }
    }
    rv[tid] = mv; ri[tid] = mi;
    __syncthreads();
    for (int st = 128; st; st >>= 1) {
      if (tid < st) {
        if (rv[tid + st] > rv[tid] ||
            (rv[tid + st] == rv[tid] && ri[tid + st] < ri[tid])) {
          rv[tid] = rv[tid + st]; ri[tid] = ri[tid + st];
        }
      }
      __syncthreads();
    }
    if (tid == 0) { s_last = ri[0]; idx[b * nS + s] = ri[0]; }
    __syncthreads();
  }
}

__global__ void gather_pos(const float* __restrict__ pos, int Ns,
                           const int* __restrict__ idx, int Nc, int Btot,
                           float* __restrict__ ctr)
{
  int t = blockIdx.x * 256 + threadIdx.x;
  if (t >= Btot * Nc * 3) return;
  int b = t / (Nc * 3), r = t - b * Nc * 3;
  int c = r / 3, d = r - c * 3;
  ctr[t] = pos[((long long)b * Ns + idx[b * Nc + c]) * 3 + d];
}

// ---------------------------------------------------------------------------
// Ball query = top_k(-where(d2<=r2,d2,inf), 64): 64 nearest-in-radius per
// center. One workgroup per center; distances in LDS; 64 min-extractions.
// ---------------------------------------------------------------------------
__global__ __launch_bounds__(256) void ballquery(const float* __restrict__ pos, int Ns,
                                                 const float* __restrict__ ctr, int Nc,
                                                 float r2, int* __restrict__ nbr,
                                                 unsigned char* __restrict__ valid)
{
  __shared__ float d2[6144];
  __shared__ float rv[256];
  __shared__ int   ri[256];
  const int b = blockIdx.x / Nc, c = blockIdx.x - b * Nc;
  const int tid = threadIdx.x;
  const float* P = pos + (long long)b * Ns * 3;
  const float cx = ctr[((long long)b * Nc + c) * 3 + 0];
  const float cy = ctr[((long long)b * Nc + c) * 3 + 1];
  const float cz = ctr[((long long)b * Nc + c) * 3 + 2];
  for (int i = tid; i < Ns; i += 256) {
    float dx = P[i * 3 + 0] - cx, dy = P[i * 3 + 1] - cy, dz = P[i * 3 + 2] - cz;
    float d = dx * dx + dy * dy + dz * dz;
    d2[i] = (d <= r2) ? d : BIGF;
  }
  __syncthreads();
  for (int k = 0; k < 64; k++) {
    float mv = BIGF; int mi = 0x7fffffff;
    for (int i = tid; i < Ns; i += 256) {
      float v = d2[i];
      if (v < mv) { mv = v; mi = i; }
    }
    rv[tid] = mv; ri[tid] = mi;
    __syncthreads();
    for (int st = 128; st; st >>= 1) {
      if (tid < st) {
        if (rv[tid + st] < rv[tid] ||
            (rv[tid + st] == rv[tid] && ri[tid + st] < ri[tid])) {
          rv[tid] = rv[tid + st]; ri[tid] = ri[tid + st];
        }
      }
      __syncthreads();
    }
    if (tid == 0) {
      long long o = ((long long)b * Nc + c) * 64 + k;
      if (rv[0] < BIGF) { nbr[o] = ri[0]; valid[o] = 1; d2[ri[0]] = BIGF; }
      else              { nbr[o] = 0;     valid[o] = 0; }
    }
    __syncthreads();
  }
}

// ---------------------------------------------------------------------------
// Gathers (all write zero-padded f16 A matrices for the WMMA GEMM).
// ---------------------------------------------------------------------------
__global__ void sa_gather(const float* __restrict__ x, const float* __restrict__ pos,
                          const float* __restrict__ ctr, const int* __restrict__ nbr,
                          int Cin, int Kp, int Nc, _Float16* __restrict__ A)
{
  int t = blockIdx.x * 256 + threadIdx.x;
  if (t >= Nc * 64 * Kp) return;
  int k = t % Kp, row = t / Kp;
  int nb = row & 63, c = row >> 6;
  int s = nbr[c * 64 + nb];
  float v = 0.f;
  if (k < Cin)          v = x[(long long)s * Cin + k];
  else if (k < Cin + 3) v = pos[s * 3 + (k - Cin)] - ctr[c * 3 + (k - Cin)];
  A[t] = (_Float16)v;
}

__global__ void sa_maxpool(const _Float16* __restrict__ H,
                           const unsigned char* __restrict__ valid,
                           int C, int Nc, float* __restrict__ X)
{
  int t = blockIdx.x * 256 + threadIdx.x;
  if (t >= Nc * C) return;
  int ch = t % C, c = t / C;
  float m = -BIGF;
  for (int nb = 0; nb < 64; nb++)
    if (valid[c * 64 + nb])
      m = fmaxf(m, (float)H[((long long)(c * 64 + nb)) * C + ch]);
  X[t] = m;
}

__global__ void cat2_gather(const float* __restrict__ f0, int C0, int s0,
                            const float* __restrict__ f1, int C1, int s1,
                            int Kp, int rows, _Float16* __restrict__ A)
{
  int t = blockIdx.x * 256 + threadIdx.x;
  if (t >= rows * Kp) return;
  int k = t % Kp, row = t / Kp;
  float v = 0.f;
  if (k < C0)           v = f0[(long long)row * s0 + k];
  else if (k < C0 + C1) v = f1[(long long)row * s1 + (k - C0)];
  A[t] = (_Float16)v;
}

__global__ void fp_gather(const float* __restrict__ fsrc, int C0,
                          const int* __restrict__ nn,
                          const float* __restrict__ xskip, int C1,
                          int Kp, int rows, _Float16* __restrict__ A)
{
  int t = blockIdx.x * 256 + threadIdx.x;
  if (t >= rows * Kp) return;
  int k = t % Kp, row = t / Kp;
  float v = 0.f;
  if (k < C0)           v = fsrc[(long long)nn[row] * C0 + k];
  else if (k < C0 + C1) v = xskip[(long long)row * C1 + (k - C0)];
  A[t] = (_Float16)v;
}

__global__ void fp1_gather(const float* __restrict__ fsrc, int C0,
                           const int* __restrict__ kI, const float* __restrict__ kW,
                           const float* __restrict__ xskip, int C1,
                           int Kp, int rows, _Float16* __restrict__ A)
{
  int t = blockIdx.x * 256 + threadIdx.x;
  if (t >= rows * Kp) return;
  int k = t % Kp, row = t / Kp;
  float v = 0.f;
  if (k < C0) {
    v = kW[row * 3 + 0] * fsrc[(long long)kI[row * 3 + 0] * C0 + k]
      + kW[row * 3 + 1] * fsrc[(long long)kI[row * 3 + 1] * C0 + k]
      + kW[row * 3 + 2] * fsrc[(long long)kI[row * 3 + 2] * C0 + k];
  } else if (k < C0 + C1) {
    v = xskip[(long long)row * C1 + (k - C0)];
  }
  A[t] = (_Float16)v;
}

__global__ void final_gather(const float* __restrict__ f1,
                             const float* __restrict__ noise,  // [32][rows]
                             int Kp, int rows, _Float16* __restrict__ A)
{
  int t = blockIdx.x * 256 + threadIdx.x;
  if (t >= rows * Kp) return;
  int k = t % Kp, row = t / Kp;
  float v;
  if (k < 128) v = f1[(long long)row * 128 + k];
  else         v = noise[(long long)(k - 128) * rows + row];
  A[t] = (_Float16)v;
}

__global__ void rowmax(const _Float16* __restrict__ H, int rows, int C,
                       float* __restrict__ g)
{
  int t = blockIdx.x * 256 + threadIdx.x;
  if (t >= C) return;
  float m = -BIGF;
  for (int r = 0; r < rows; r++)
    m = fmaxf(m, (float)H[(long long)r * C + t]);
  g[t] = m;
}

__global__ void knn1(const float* __restrict__ src, int Ns,
                     const float* __restrict__ tgt, int Nt, int Btot,
                     int* __restrict__ nn)
{
  int t = blockIdx.x * 256 + threadIdx.x;
  if (t >= Btot * Nt) return;
  int b = t / Nt;
  const float* S = src + (long long)b * Ns * 3;
  float tx = tgt[t * 3 + 0], ty = tgt[t * 3 + 1], tz = tgt[t * 3 + 2];
  float bd = BIGF; int bi = 0;
  for (int j = 0; j < Ns; j++) {
    float dx = tx - S[j * 3 + 0], dy = ty - S[j * 3 + 1], dz = tz - S[j * 3 + 2];
    float d = dx * dx + dy * dy + dz * dz;
    if (d < bd) { bd = d; bi = j; }
  }
  nn[t] = bi;
}

__global__ void knn3(const float* __restrict__ src, int Ns,
                     const float* __restrict__ tgt, int Nt, int Btot,
                     int* __restrict__ nnI, float* __restrict__ nnW)
{
  int t = blockIdx.x * 256 + threadIdx.x;
  if (t >= Btot * Nt) return;
  int b = t / Nt;
  const float* S = src + (long long)b * Ns * 3;
  float tx = tgt[t * 3 + 0], ty = tgt[t * 3 + 1], tz = tgt[t * 3 + 2];
  float d0 = BIGF, d1 = BIGF, d2 = BIGF; int i0 = 0, i1 = 0, i2 = 0;
  for (int j = 0; j < Ns; j++) {
    float dx = tx - S[j * 3 + 0], dy = ty - S[j * 3 + 1], dz = tz - S[j * 3 + 2];
    float d = dx * dx + dy * dy + dz * dz;
    if (d < d0)      { d2 = d1; i2 = i1; d1 = d0; i1 = i0; d0 = d; i0 = j; }
    else if (d < d1) { d2 = d1; i2 = i1; d1 = d;  i1 = j; }
    else if (d < d2) { d2 = d;  i2 = j; }
  }
  float w0 = 1.f / (d0 + 1e-16f), w1 = 1.f / (d1 + 1e-16f), w2 = 1.f / (d2 + 1e-16f);
  float inv = 1.f / (w0 + w1 + w2);
  nnI[t * 3 + 0] = i0; nnI[t * 3 + 1] = i1; nnI[t * 3 + 2] = i2;
  nnW[t * 3 + 0] = w0 * inv; nnW[t * 3 + 1] = w1 * inv; nnW[t * 3 + 2] = w2 * inv;
}

// ---------------------------------------------------------------------------
// Host orchestration
// ---------------------------------------------------------------------------
struct Lay {
  const float *W, *b, *g, *be;
  int Kin, Nout, Kp, Np, epi;
  _Float16* Wt;
};

extern "C" void kernel_launch(void* const* d_in, const int* in_sizes, int n_in,
                              void* d_out, int out_size, void* d_ws, size_t ws_size,
                              hipStream_t stream)
{
  (void)in_sizes; (void)n_in; (void)out_size; (void)ws_size;
  const int Bn = 16, N0 = 6144;

  const float* cloud = (const float*)d_in[0];
  const float* noise = (const float*)d_in[1];
  float* out = (float*)d_out;

  // params flattened by jax tree (dicts in sorted-key order):
  // W1,W2,W3,b1,b2,b3,be1,be2, fp1, fp2, fp3, fp4, g1,g2, sa1, sa2, sa3, sa4
  // each MLP layer contributes W,b[,be,g] (sorted: 'W' < 'b' < 'be' < 'g').
  int ti = 2;
  auto NX = [&]() { return (const float*)d_in[ti++]; };
  auto parseChain = [&](Lay* L, int n, const int* d) {
    for (int i = 0; i < n; i++) {
      L[i].W = NX(); L[i].b = NX();
      if (i < n - 1) { L[i].be = NX(); L[i].g = NX(); L[i].epi = EPI_LNRELU; }
      else           { L[i].be = nullptr; L[i].g = nullptr; L[i].epi = EPI_NONE; }
      L[i].Kin = d[i]; L[i].Nout = d[i + 1];
      L[i].Kp = (d[i] + 31) / 32 * 32;
      L[i].Np = (d[i + 1] + 15) / 16 * 16;
      L[i].Wt = nullptr;
    }
  };

  const float *W1 = NX(), *W2 = NX(), *W3 = NX();
  const float *b1 = NX(), *b2 = NX(), *b3 = NX();
  const float *be1 = NX(), *be2 = NX();

  static const int dfp1[] = {131, 128, 128, 128};
  static const int dfp2[] = {384, 256, 128};
  static const int dfp3[] = {768, 512, 256};
  static const int dfp4[] = {1536, 512, 512};
  static const int dsa1[] = {6, 64, 64, 128};
  static const int dsa2[] = {131, 128, 128, 256};
  static const int dsa3[] = {259, 256, 256, 512};
  static const int dsa4[] = {515, 256, 512, 1024};

  Lay fp1L[3]; parseChain(fp1L, 3, dfp1);
  Lay fp2L[2]; parseChain(fp2L, 2, dfp2);
  Lay fp3L[2]; parseChain(fp3L, 2, dfp3);
  Lay fp4L[2]; parseChain(fp4L, 2, dfp4);
  const float *g1 = NX(), *g2 = NX();
  Lay sa1L[3]; parseChain(sa1L, 3, dsa1);
  Lay sa2L[3]; parseChain(sa2L, 3, dsa2);
  Lay sa3L[3]; parseChain(sa3L, 3, dsa3);
  Lay sa4L[3]; parseChain(sa4L, 3, dsa4);

  Lay finL[3];
  finL[0].W = W1; finL[0].b = b1; finL[0].g = g1; finL[0].be = be1;
  finL[0].Kin = 160; finL[0].Nout = 128; finL[0].Kp = 160; finL[0].Np = 128; finL[0].epi = EPI_LN;
  finL[1].W = W2; finL[1].b = b2; finL[1].g = g2; finL[1].be = be2;
  finL[1].Kin = 128; finL[1].Nout = 64; finL[1].Kp = 128; finL[1].Np = 64; finL[1].epi = EPI_LN;
  finL[2].W = W3; finL[2].b = b3; finL[2].g = nullptr; finL[2].be = nullptr;
  finL[2].Kin = 64; finL[2].Nout = 3; finL[2].Kp = 64; finL[2].Np = 16; finL[2].epi = EPI_SIG;

  // ---- workspace bump allocator ----
  size_t woff = 0;
  auto alloc = [&](size_t bytes) -> void* {
    woff = (woff + 255) & ~(size_t)255;
    void* p = (char*)d_ws + woff;
    woff += bytes;
    return p;
  };
  auto cdiv = [](int a, int b) { return (a + b - 1) / b; };

  // convert all weights to padded/transposed f16
  Lay* chains[] = {fp1L, fp2L, fp3L, fp4L, sa1L, sa2L, sa3L, sa4L, finL};
  const int chN[] = {3, 2, 2, 2, 3, 3, 3, 3, 3};
  for (int c = 0; c < 9; c++)
    for (int l = 0; l < chN[c]; l++) {
      Lay& L = chains[c][l];
      L.Wt = (_Float16*)alloc((size_t)L.Np * L.Kp * sizeof(_Float16));
      convert_wt<<<cdiv(L.Np * L.Kp, 256), 256, 0, stream>>>(L.W, L.Kin, L.Nout, L.Np, L.Kp, L.Wt);
    }

  // big ping-pong f16 buffers (per-batch), sized for sa1 L3 out (24576 x 128)
  _Float16* bufA = (_Float16*)alloc((size_t)24576 * 128 * sizeof(_Float16));
  _Float16* bufB = (_Float16*)alloc((size_t)24576 * 128 * sizeof(_Float16));

  // position-derived full-batch buffers
  float* pos1 = (float*)alloc((size_t)Bn * 384 * 3 * 4);
  float* pos2 = (float*)alloc((size_t)Bn * 64 * 3 * 4);
  float* pos3 = (float*)alloc((size_t)Bn * 32 * 3 * 4);
  int* idx1 = (int*)alloc((size_t)Bn * 384 * 4);
  int* idx2 = (int*)alloc((size_t)Bn * 64 * 4);
  int* idx3 = (int*)alloc((size_t)Bn * 32 * 4);
  int* nbr1 = (int*)alloc((size_t)Bn * 384 * 64 * 4);
  int* nbr2 = (int*)alloc((size_t)Bn * 64 * 64 * 4);
  int* nbr3 = (int*)alloc((size_t)Bn * 32 * 64 * 4);
  unsigned char* val1 = (unsigned char*)alloc((size_t)Bn * 384 * 64);
  unsigned char* val2 = (unsigned char*)alloc((size_t)Bn * 64 * 64);
  unsigned char* val3 = (unsigned char*)alloc((size_t)Bn * 32 * 64);
  int* nn3 = (int*)alloc((size_t)Bn * 64 * 4);
  int* nn2 = (int*)alloc((size_t)Bn * 384 * 4);
  int* kI = (int*)alloc((size_t)Bn * N0 * 3 * 4);
  float* kW = (float*)alloc((size_t)Bn * N0 * 3 * 4);

  // per-batch feature scratch (f32)
  float* x1 = (float*)alloc((size_t)384 * 128 * 4);
  float* x2 = (float*)alloc((size_t)64 * 256 * 4);
  float* x3 = (float*)alloc((size_t)32 * 512 * 4);
  float* gb = (float*)alloc((size_t)1024 * 4);
  float* f4 = (float*)alloc((size_t)32 * 512 * 4);
  float* f3 = (float*)alloc((size_t)64 * 256 * 4);
  float* f2 = (float*)alloc((size_t)384 * 128 * 4);
  float* f1 = (float*)alloc((size_t)N0 * 128 * 4);

  auto lin = [&](const Lay& L, int rows, const _Float16* Ain, _Float16* Ho,
                 float* Fo, int epi) {
    lin_kernel<<<dim3(rows / 16), dim3(256), (size_t)16 * L.Np * 4, stream>>>(
        Ain, L.Kp, L.Wt, L.b, L.g, L.be, Ho, Fo, L.Np, L.Nout, epi);
  };

  // ---- phase 1: positions (full batch parallel) ----
  fps_kernel<<<Bn, 256, 0, stream>>>(cloud, N0, 384, idx1);
  gather_pos<<<cdiv(Bn * 384 * 3, 256), 256, 0, stream>>>(cloud, N0, idx1, 384, Bn, pos1);
  fps_kernel<<<Bn, 256, 0, stream>>>(pos1, 384, 64, idx2);
  gather_pos<<<cdiv(Bn * 64 * 3, 256), 256, 0, stream>>>(pos1, 384, idx2, 64, Bn, pos2);
  fps_kernel<<<Bn, 256, 0, stream>>>(pos2, 64, 32, idx3);
  gather_pos<<<cdiv(Bn * 32 * 3, 256), 256, 0, stream>>>(pos2, 64, idx3, 32, Bn, pos3);

  ballquery<<<Bn * 384, 256, 0, stream>>>(cloud, N0, pos1, 384, 0.1f * 0.1f, nbr1, val1);
  ballquery<<<Bn * 64, 256, 0, stream>>>(pos1, 384, pos2, 64, 0.2f * 0.2f, nbr2, val2);
  ballquery<<<Bn * 32, 256, 0, stream>>>(pos2, 64, pos3, 32, 0.4f * 0.4f, nbr3, val3);

  knn1<<<cdiv(Bn * 64, 256), 256, 0, stream>>>(pos3, 32, pos2, 64, Bn, nn3);
  knn1<<<cdiv(Bn * 384, 256), 256, 0, stream>>>(pos2, 64, pos1, 384, Bn, nn2);
  knn3<<<cdiv(Bn * N0, 256), 256, 0, stream>>>(pos1, 384, cloud, N0, Bn, kI, kW);

  // ---- phase 2: per-batch feature pipeline ----
  for (int b = 0; b < Bn; b++) {
    const float* cl = cloud + (size_t)b * N0 * 3;
    const float* p1 = pos1 + (size_t)b * 384 * 3;
    const float* p2 = pos2 + (size_t)b * 64 * 3;
    const float* p3 = pos3 + (size_t)b * 32 * 3;

    // SA1: 384 centers x 64 nbrs, feats 6 -> [64,64,128], maxpool -> x1
    sa_gather<<<cdiv(24576 * 32, 256), 256, 0, stream>>>(cl, cl, p1, nbr1 + b * 384 * 64, 3, 32, 384, bufA);
    lin(sa1L[0], 24576, bufA, bufB, nullptr, EPI_LNRELU);
    lin(sa1L[1], 24576, bufB, bufA, nullptr, EPI_LNRELU);
    lin(sa1L[2], 24576, bufA, bufB, nullptr, EPI_NONE);
    sa_maxpool<<<cdiv(384 * 128, 256), 256, 0, stream>>>(bufB, val1 + b * 384 * 64, 128, 384, x1);

    // SA2: 64 x 64, feats 131 -> [128,128,256] -> x2
    sa_gather<<<cdiv(4096 * 160, 256), 256, 0, stream>>>(x1, p1, p2, nbr2 + b * 64 * 64, 128, 160, 64, bufA);
    lin(sa2L[0], 4096, bufA, bufB, nullptr, EPI_LNRELU);
    lin(sa2L[1], 4096, bufB, bufA, nullptr, EPI_LNRELU);
    lin(sa2L[2], 4096, bufA, bufB, nullptr, EPI_NONE);
    sa_maxpool<<<cdiv(64 * 256, 256), 256, 0, stream>>>(bufB, val2 + b * 64 * 64, 256, 64, x2);

    // SA3: 32 x 64, feats 259 -> [256,256,512] -> x3
    sa_gather<<<cdiv(2048 * 288, 256), 256, 0, stream>>>(x2, p2, p3, nbr3 + b * 32 * 64, 256, 288, 32, bufA);
    lin(sa3L[0], 2048, bufA, bufB, nullptr, EPI_LNRELU);
    lin(sa3L[1], 2048, bufB, bufA, nullptr, EPI_LNRELU);
    lin(sa3L[2], 2048, bufA, bufB, nullptr, EPI_NONE);
    sa_maxpool<<<cdiv(32 * 512, 256), 256, 0, stream>>>(bufB, val3 + b * 32 * 64, 512, 32, x3);

    // SA4: concat(x3,pos3)=515 -> [256,512,1024], global max -> g
    cat2_gather<<<cdiv(32 * 544, 256), 256, 0, stream>>>(x3, 512, 512, p3, 3, 3, 544, 32, bufA);
    lin(sa4L[0], 32, bufA, bufB, nullptr, EPI_LNRELU);
    lin(sa4L[1], 32, bufB, bufA, nullptr, EPI_LNRELU);
    lin(sa4L[2], 32, bufA, bufB, nullptr, EPI_NONE);
    rowmax<<<cdiv(1024, 256), 256, 0, stream>>>(bufB, 32, 1024, gb);

    // FP4: concat(g bcast 1024, x3 512) -> [512,512] -> f4
    cat2_gather<<<cdiv(32 * 1536, 256), 256, 0, stream>>>(gb, 1024, 0, x3, 512, 512, 1536, 32, bufA);
    lin(fp4L[0], 32, bufA, bufB, nullptr, EPI_LNRELU);
    lin(fp4L[1], 32, bufB, nullptr, f4, EPI_NONE);

    // FP3: nn(pos3->pos2), concat(f4[nn] 512, x2 256) -> [512,256] -> f3
    fp_gather<<<cdiv(64 * 768, 256), 256, 0, stream>>>(f4, 512, nn3 + b * 64, x2, 256, 768, 64, bufA);
    lin(fp3L[0], 64, bufA, bufB, nullptr, EPI_LNRELU);
    lin(fp3L[1], 64, bufB, nullptr, f3, EPI_NONE);

    // FP2: nn(pos2->pos1), concat(f3[nn] 256, x1 128) -> [256,128] -> f2
    fp_gather<<<cdiv(384 * 384, 256), 256, 0, stream>>>(f3, 256, nn2 + b * 384, x1, 128, 384, 384, bufA);
    lin(fp2L[0], 384, bufA, bufB, nullptr, EPI_LNRELU);
    lin(fp2L[1], 384, bufB, nullptr, f2, EPI_NONE);

    // FP1: 3-NN interp (pos1->pos0), concat(xi 128, cloud 3) -> [128,128,128] -> f1
    fp1_gather<<<cdiv(N0 * 160, 256), 256, 0, stream>>>(f2, 128, kI + b * N0 * 3, kW + b * N0 * 3,
                                                        cl, 3, 160, N0, bufA);
    lin(fp1L[0], N0, bufA, bufB, nullptr, EPI_LNRELU);
    lin(fp1L[1], N0, bufB, bufA, nullptr, EPI_LNRELU);
    lin(fp1L[2], N0, bufA, nullptr, f1, EPI_NONE);

    // Final: concat(f1 128, noise^T 32)=160 -> LN -> LN -> sigmoid*2-1 -> d_out
    final_gather<<<cdiv(N0 * 160, 256), 256, 0, stream>>>(f1, noise + (size_t)b * 32 * N0, 160, N0, bufA);
    lin(finL[0], N0, bufA, bufB, nullptr, EPI_LN);
    lin(finL[1], N0, bufB, bufA, nullptr, EPI_LN);
    lin(finL[2], N0, bufA, nullptr, out + (size_t)b * N0 * 3, EPI_SIG);
  }
}
